// Net_63531156242921
// MI455X (gfx1250) — matrix operand order repaired and checked by reading, not compiled, not run on hardware
//
#include <hip/hip_runtime.h>
#include <hip/hip_bf16.h>
#include <stdint.h>

#define T_STEPS 50
#define BSZ     128
#define DIN     768
#define HID     2048
#define NOUT    10
#define NP4     16   // N=10 padded to one WMMA tile

#define BN      64   // N columns per block (big layers)
#define LDS_STRIDE 72  // halfs per K-row in LDS (padded: 36-bank stride, conflict-free)

// scheduling-group hints (no-op if builtin unavailable)
#if defined(__has_builtin)
#if __has_builtin(__builtin_amdgcn_sched_group_barrier)
#define SGB(mask, cnt) __builtin_amdgcn_sched_group_barrier((mask), (cnt), 0)
#endif
#endif
#ifndef SGB
#define SGB(mask, cnt)
#endif

typedef __attribute__((ext_vector_type(8)))  _Float16 v8h;
typedef __attribute__((ext_vector_type(16))) _Float16 v16h;
typedef __attribute__((ext_vector_type(8)))  float    v8f;

union HFrag { v16h v; v8h h[2]; };

// ---------------- utility kernels ----------------

__global__ void k_zero(uint32_t* __restrict__ p, long n) {
  long i = (long)blockIdx.x * blockDim.x + threadIdx.x;
  if (i < n) p[i] = 0u;
}

// x: [B, T, DIN] f32 -> xh: [T, B, DIN] f16
__global__ void k_conv_x(const float* __restrict__ x, _Float16* __restrict__ xh) {
  const long n = (long)T_STEPS * BSZ * DIN;
  long i = (long)blockIdx.x * blockDim.x + threadIdx.x;
  if (i >= n) return;
  int d = (int)(i % DIN);
  long r = i / DIN;
  int b = (int)(r % BSZ);
  int t = (int)(r / BSZ);
  xh[i] = (_Float16)x[((long)b * T_STEPS + t) * DIN + d];
}

// W: [N,K] f32 -> Wt: [K,N] f16   (B-operand wants K rows contiguous in N)
__global__ void k_transpose_h(const float* __restrict__ W, _Float16* __restrict__ Wt,
                              int N, int K) {
  const long n = (long)N * K;
  long i = (long)blockIdx.x * blockDim.x + threadIdx.x;
  if (i >= n) return;
  int nn = (int)(i % N);
  int k  = (int)(i / N);
  Wt[i] = (_Float16)W[(long)nn * K + k];
}

// W4: [10, HID] f32 -> Wt4: [HID, 16] f16 zero-padded; b4 -> b4p[16]
__global__ void k_w4(const float* __restrict__ W4, const float* __restrict__ b4,
                     _Float16* __restrict__ Wt4, float* __restrict__ b4p) {
  const long n = (long)HID * NP4;
  long i = (long)blockIdx.x * blockDim.x + threadIdx.x;
  if (i < NP4) b4p[i] = (i < NOUT) ? b4[i] : 0.0f;
  if (i >= n) return;
  int nn = (int)(i % NP4);
  int k  = (int)(i / NP4);
  Wt4[i] = (nn < NOUT) ? (_Float16)W4[(long)nn * HID + k] : (_Float16)0.0f;
}

// ---------------- LIF epilogue (shared) ----------------
// mem' = clip(beta,0,1)*mem + (acc + bias) - spk_prev*thr ;  spk = (mem' - thr > 0)
__device__ __forceinline__ void lif_store(
    float accv, float bia, float beta, float thr,
    float* __restrict__ mem, _Float16* __restrict__ spkst,
    float* __restrict__ spk_out, float* __restrict__ mem_out,
    int m, int n, int Npad, int Nlog)
{
  const float cur   = accv + bia;
  const size_t sidx = (size_t)m * Npad + n;
  const float mprev = mem[sidx];
  const float sprev = (float)spkst[sidx];
  const float mnew  = beta * mprev + cur - sprev * thr;
  const float s     = (mnew > thr) ? 1.0f : 0.0f;
  mem[sidx]   = mnew;
  spkst[sidx] = (_Float16)s;
  if (n < Nlog) {
    const size_t o = (size_t)m * Nlog + n;
    __builtin_nontemporal_store(s,    spk_out + o);  // keep 315MB of outputs out of L2
    __builtin_nontemporal_store(mnew, mem_out + o);
  }
}

// ---------------- big-layer GEMM: LDS-staged B, 128x64 block tile ----------------
// Block owns all 128 rows x 64 cols. Wave w computes rows [16w,16w+16) x 64 cols
// (4 WMMA accumulators, A fragment reused 4x). B chunk (32K x 64N f16) staged in
// LDS double-buffered through registers; A fragment software-pipelined one chunk
// ahead. Prefetch indices are clamped so the loop body is branch-free (single
// scheduling region) and sched_group_barriers pin the issue order:
//   VMEM_READ x3 -> DS_READ x8 -> WMMA x4 -> DS_WRITE x1.
__global__ __launch_bounds__(256) void k_gemm_lif_lds(
    const _Float16* __restrict__ A,    // [128, K] row-major
    const _Float16* __restrict__ Wt,   // [K, HID] row-major (pre-transposed weight)
    const float*    __restrict__ bias, // [HID]
    float*          __restrict__ mem,  // [128, HID] state
    _Float16*       __restrict__ spkst,// [128, HID] prev-spike state (f16, exact 0/1)
    float*          __restrict__ spk_out,  // [128, HID]
    float*          __restrict__ mem_out,  // [128, HID]
    const float* __restrict__ beta_p, const float* __restrict__ thr_p,
    int K)
{
  __shared__ __align__(16) _Float16 Bs[2][32 * LDS_STRIDE];  // 9216 B of 320KB/WGP

  const int tid  = threadIdx.x;
  const int lane = tid & 31;
  const int wave = tid >> 5;                      // 0..7 -> M tile
  const int m0   = wave * 16;
  const int n0   = blockIdx.x * BN;
  const int hi   = (lane >> 4) * 8;                    // A K-chunk select (ISA 7.12.2)
  const int kln  = (lane & 15) + ((lane >> 4) << 4);   // B K-row per lane (0..31)

  // staging assignment: thread -> one 16B piece of the 32x64 B chunk
  const int sr = tid >> 3;        // K row 0..31
  const int sc = (tid & 7) * 8;   // N col 0,8,..,56

  const int nchunks = K / 32;

  // prologue: stage B chunk 0 into LDS, A chunk 0 into registers
  {
    v8h stg = *(const v8h*)(Wt + (size_t)sr * HID + n0 + sc);
    *(v8h*)(&Bs[0][sr * LDS_STRIDE + sc]) = stg;
  }
  const _Float16* Ap = A + (size_t)(m0 + (lane & 15)) * K;
  HFrag a_cur;
  a_cur.h[0] = *(const v8h*)(Ap + hi);          // K = {0..7 | 8..15}
  a_cur.h[1] = *(const v8h*)(Ap + 16 + hi);     // K = {16..23 | 24..31}
  __syncthreads();

  v8f acc[4] = {};

  for (int kc = 0; kc < nchunks; ++kc) {
    const int cur = kc & 1;
    // clamped prefetch index: last iteration harmlessly re-reads the final chunk,
    // keeping the body branch-free
    const int knx = (kc + 1 < nchunks) ? (kc + 1) : kc;

    // next chunk's global loads (retire under this chunk's WMMAs)
    v8h stg = *(const v8h*)(Wt + ((size_t)knx * 32 + sr) * HID + n0 + sc);
    HFrag a_nxt;
    a_nxt.h[0] = *(const v8h*)(Ap + knx * 32 + hi);
    a_nxt.h[1] = *(const v8h*)(Ap + knx * 32 + 16 + hi);

    // all 8 ds_loads for the 4 B fragments, then 4 WMMAs (progressive dscnt waits)
    const _Float16* Bl = &Bs[cur][hi * LDS_STRIDE + (lane & 15)];
    HFrag b[4];
#pragma unroll
    for (int j = 0; j < 4; ++j) {
#pragma unroll
      for (int s = 0; s < 16; ++s)               // K = {0..7 | 8..15}, {16..23 | 24..31}; N = n0+j*16+lane%16
        b[j].v[s] = Bl[(s < 8 ? s : s + 8) * LDS_STRIDE + j * 16];
    }
#pragma unroll
    for (int j = 0; j < 4; ++j) {
      acc[j] = __builtin_amdgcn_wmma_f32_16x16x32_f16(
          false, a_cur.v, false, b[j].v, (short)0, acc[j], false, false);
    }

    *(v8h*)(&Bs[1 - cur][sr * LDS_STRIDE + sc]) = stg;
    a_cur = a_nxt;

    SGB(0x020, 3);   // VMEM reads first (longest latency)
    SGB(0x100, 8);   // then all 8 LDS reads
    SGB(0x008, 4);   // then the 4 WMMAs
    SGB(0x200, 1);   // then the LDS write of the staged chunk
    __syncthreads();
  }

  float beta = *beta_p;
  beta = beta < 0.f ? 0.f : (beta > 1.f ? 1.f : beta);
  const float thr = *thr_p;

#pragma unroll
  for (int j = 0; j < 4; ++j) {
    const int n = n0 + j * 16 + (lane & 15);
    const float bia = bias[n];
#pragma unroll
    for (int r = 0; r < 8; ++r) {
      const int m = m0 + r + hi;                 // C/D: lanes>=16 hold M=r+8
      lif_store(acc[j][r], bia, beta, thr, mem, spkst, spk_out, mem_out,
                m, n, HID, HID);
    }
  }
}

// ---------------- small-layer GEMM (layer 4, N padded to 16): direct global ----------------
__global__ __launch_bounds__(256) void k_gemm_lif_small(
    const _Float16* __restrict__ A,    // [128, K]
    const _Float16* __restrict__ Wt,   // [K, NP4]
    const float*    __restrict__ bias, // [NP4]
    float*          __restrict__ mem,  // [128, NP4]
    _Float16*       __restrict__ spkst,// [128, NP4]
    float*          __restrict__ spk_out,  // [128, NOUT]
    float*          __restrict__ mem_out,  // [128, NOUT]
    const float* __restrict__ beta_p, const float* __restrict__ thr_p,
    int K)
{
  const int lane = threadIdx.x & 31;
  const int wave = threadIdx.x >> 5;             // 0..7 -> M tile
  const int m0   = wave * 16;
  const int hi   = (lane >> 4) * 8;
  const int kln  = (lane & 15) + ((lane >> 4) << 4);

  v8f acc = {};
  const _Float16* Ap = A + (size_t)(m0 + (lane & 15)) * K;
  for (int k0 = 0; k0 < K; k0 += 32) {
    HFrag a;
    a.h[0] = *(const v8h*)(Ap + k0 + hi);
    a.h[1] = *(const v8h*)(Ap + k0 + 16 + hi);
    const _Float16* Bp = Wt + (size_t)(k0 + hi) * NP4 + (lane & 15);
    HFrag b;
#pragma unroll
    for (int s = 0; s < 16; ++s)
      b.v[s] = Bp[(s < 8 ? s : s + 8) * NP4];
    acc = __builtin_amdgcn_wmma_f32_16x16x32_f16(
        false, a.v, false, b.v, (short)0, acc, false, false);
  }

  float beta = *beta_p;
  beta = beta < 0.f ? 0.f : (beta > 1.f ? 1.f : beta);
  const float thr = *thr_p;

  const int n = lane & 15;
  const float bia = bias[n];
#pragma unroll
  for (int r = 0; r < 8; ++r) {
    const int m = m0 + r + hi;
    lif_store(acc[r], bia, beta, thr, mem, spkst, spk_out, mem_out,
              m, n, NP4, NOUT);
  }
}

// ---------------- host sequencing ----------------

extern "C" void kernel_launch(void* const* d_in, const int* in_sizes, int n_in,
                              void* d_out, int out_size, void* d_ws, size_t ws_size,
                              hipStream_t stream) {
  (void)in_sizes; (void)n_in; (void)out_size; (void)ws_size;
  const float* x  = (const float*)d_in[0];
  const float* W1 = (const float*)d_in[1];
  const float* b1 = (const float*)d_in[2];
  const float* W2 = (const float*)d_in[3];
  const float* b2 = (const float*)d_in[4];
  const float* W3 = (const float*)d_in[5];
  const float* b3 = (const float*)d_in[6];
  const float* W4 = (const float*)d_in[7];
  const float* b4 = (const float*)d_in[8];
  const float* beta1 = (const float*)d_in[9];
  const float* beta2 = (const float*)d_in[10];
  const float* beta3 = (const float*)d_in[11];
  const float* beta4 = (const float*)d_in[12];
  const float* thr1  = (const float*)d_in[13];
  const float* thr2  = (const float*)d_in[14];
  const float* thr3  = (const float*)d_in[15];
  const float* thr4  = (const float*)d_in[16];

  char* ws = (char*)d_ws;
  size_t off = 0;
  auto alloc = [&](size_t bytes) -> char* {
    char* p = ws + off;
    off += bytes;
    off = (off + 255) & ~(size_t)255;
    return p;
  };

  _Float16* xh  = (_Float16*)alloc((size_t)T_STEPS * BSZ * DIN * 2);
  _Float16* W1t = (_Float16*)alloc((size_t)DIN * HID * 2);
  _Float16* W2t = (_Float16*)alloc((size_t)HID * HID * 2);
  _Float16* W3t = (_Float16*)alloc((size_t)HID * HID * 2);
  _Float16* W4t = (_Float16*)alloc((size_t)HID * NP4 * 2);
  float*    b4p = (float*)alloc((size_t)NP4 * 4);

  char* state0 = ws + off;
  float*    mem1 = (float*)alloc((size_t)BSZ * HID * 4);
  float*    mem2 = (float*)alloc((size_t)BSZ * HID * 4);
  float*    mem3 = (float*)alloc((size_t)BSZ * HID * 4);
  float*    mem4 = (float*)alloc((size_t)BSZ * NP4 * 4);
  _Float16* s1h  = (_Float16*)alloc((size_t)BSZ * HID * 2);
  _Float16* s2h  = (_Float16*)alloc((size_t)BSZ * HID * 2);
  _Float16* s3h  = (_Float16*)alloc((size_t)BSZ * HID * 2);
  _Float16* s4h  = (_Float16*)alloc((size_t)BSZ * NP4 * 2);
  const long stateWords = (long)(((ws + off) - state0) / 4);

  // 1) zero LIF state every call (deterministic)
  k_zero<<<(int)((stateWords + 255) / 256), 256, 0, stream>>>((uint32_t*)state0, stateWords);

  // 2) per-call precision/layout conversions, amortized over 50 steps
  { long n = (long)T_STEPS * BSZ * DIN;
    k_conv_x<<<(int)((n + 255) / 256), 256, 0, stream>>>(x, xh); }
  { long n = (long)DIN * HID;
    k_transpose_h<<<(int)((n + 255) / 256), 256, 0, stream>>>(W1, W1t, HID, DIN); }
  { long n = (long)HID * HID;
    k_transpose_h<<<(int)((n + 255) / 256), 256, 0, stream>>>(W2, W2t, HID, HID);
    k_transpose_h<<<(int)((n + 255) / 256), 256, 0, stream>>>(W3, W3t, HID, HID); }
  { long n = (long)HID * NP4;
    k_w4<<<(int)((n + 255) / 256), 256, 0, stream>>>(W4, b4, W4t, b4p); }

  // 3) output slices: (s1,s2,s3,s4,m1,m2,m3,m4) each stacked over T
  float* out = (float*)d_out;
  const size_t szh = (size_t)T_STEPS * BSZ * HID;
  const size_t szo = (size_t)T_STEPS * BSZ * NOUT;
  float* s1o = out;
  float* s2o = s1o + szh;
  float* s3o = s2o + szh;
  float* s4o = s3o + szh;
  float* m1o = s4o + szo;
  float* m2o = m1o + szh;
  float* m3o = m2o + szh;
  float* m4o = m3o + szh;

  const int blocksH = HID / BN;  // 32 blocks x 8 waves, each block owns 128x64

  for (int t = 0; t < T_STEPS; ++t) {
    const _Float16* xt = xh + (size_t)t * BSZ * DIN;
    float* so; float* mo;

    so = s1o + (size_t)t * BSZ * HID; mo = m1o + (size_t)t * BSZ * HID;
    k_gemm_lif_lds<<<blocksH, 256, 0, stream>>>(xt, W1t, b1, mem1, s1h, so, mo,
                                                beta1, thr1, DIN);

    so = s2o + (size_t)t * BSZ * HID; mo = m2o + (size_t)t * BSZ * HID;
    k_gemm_lif_lds<<<blocksH, 256, 0, stream>>>(s1h, W2t, b2, mem2, s2h, so, mo,
                                                beta2, thr2, HID);

    so = s3o + (size_t)t * BSZ * HID; mo = m3o + (size_t)t * BSZ * HID;
    k_gemm_lif_lds<<<blocksH, 256, 0, stream>>>(s2h, W3t, b3, mem3, s3h, so, mo,
                                                beta3, thr3, HID);

    so = s4o + (size_t)t * BSZ * NOUT; mo = m4o + (size_t)t * BSZ * NOUT;
    k_gemm_lif_small<<<1, 256, 0, stream>>>(s3h, W4t, b4p, mem4, s4h, so, mo,
                                            beta4, thr4, HID);
  }
}
